// ConvCNP_37847251812836
// MI455X (gfx1250) — matrix operand order repaired and checked by reading, not compiled, not run on hardware
//
#include <hip/hip_runtime.h>
#include <hip/hip_bf16.h>
#include <math.h>

#define GRES 128
#define BATCH 8
#define NC 1024
#define NT 2048
#define HID 128

typedef __bf16 bf16_t;
typedef __attribute__((ext_vector_type(16))) __bf16 v16bf;
typedef __attribute__((ext_vector_type(8)))  __bf16 v8bf;
typedef __attribute__((ext_vector_type(8)))  float  v8f;

union ABFrag { v16bf v; v8bf h[2]; };

// ---------------------------------------------------------------------------
// 1) RBF SetConv encoder: density + normalized signal on the 128x128 grid.
// feat layout: [B][H][W][2] fp32  (H ~ y/row, W ~ x/col, matching reference)
// ---------------------------------------------------------------------------
__global__ __launch_bounds__(256)
void rbf_kernel(const float* __restrict__ xc, const float* __restrict__ yc,
                const float* __restrict__ sigma_p, float* __restrict__ feat) {
    __shared__ float sx[256], sy[256], sv[256];
    const int b = blockIdx.y;
    const int g = blockIdx.x * 256 + threadIdx.x;
    const int row = g >> 7;      // y index (dim i)
    const int col = g & 127;     // x index (dim j)
    const float gx = -1.f + 2.f * (float)col / (float)(GRES - 1);
    const float gy = -1.f + 2.f * (float)row / (float)(GRES - 1);
    const float sig = sigma_p[0];
    const float coef = -0.5f / (sig * sig);
    float density = 0.f, signal = 0.f;
    for (int c0 = 0; c0 < NC; c0 += 256) {
        const int ci = c0 + threadIdx.x;
        sx[threadIdx.x] = xc[((size_t)b * NC + ci) * 2 + 0];
        sy[threadIdx.x] = xc[((size_t)b * NC + ci) * 2 + 1];
        sv[threadIdx.x] = yc[(size_t)b * NC + ci];
        __syncthreads();
#pragma unroll 4
        for (int k = 0; k < 256; ++k) {
            const float dx = gx - sx[k];
            const float dy = gy - sy[k];
            const float w = __expf(coef * (dx * dx + dy * dy));
            density += w;
            signal  += w * sv[k];
        }
        __syncthreads();
    }
    float* f = feat + ((size_t)b * GRES * GRES + g) * 2;
    f[0] = density;
    f[1] = signal / (density + 1e-5f);
}

// ---------------------------------------------------------------------------
// 2) conv1: 2 -> 128, 3x3, ReLU. K is tiny (18) so direct VALU.
// Output h1: [B][H][W][128] bf16 (channel-fastest for the WMMA convs).
// ---------------------------------------------------------------------------
__global__ __launch_bounds__(256)
void conv1_kernel(const float* __restrict__ feat, const float* __restrict__ w1,
                  const float* __restrict__ b1, bf16_t* __restrict__ h1) {
    const int b = blockIdx.y;
    const int g = blockIdx.x * 256 + threadIdx.x;
    const int row = g >> 7, col = g & 127;
    float patch[2][3][3];
#pragma unroll
    for (int ky = 0; ky < 3; ++ky)
#pragma unroll
        for (int kx = 0; kx < 3; ++kx) {
            const int y = row + ky - 1, x = col + kx - 1;
            const bool ok = ((unsigned)y < GRES) && ((unsigned)x < GRES);
            const float* p = feat + (((size_t)b * GRES + (ok ? y : 0)) * GRES + (ok ? x : 0)) * 2;
            patch[0][ky][kx] = ok ? p[0] : 0.f;
            patch[1][ky][kx] = ok ? p[1] : 0.f;
        }
    bf16_t* out = h1 + ((size_t)b * GRES * GRES + g) * HID;
    for (int oc = 0; oc < HID; ++oc) {
        float acc = b1[oc];
        const float* w = w1 + (size_t)oc * 2 * 9;
#pragma unroll
        for (int ic = 0; ic < 2; ++ic)
#pragma unroll
            for (int ky = 0; ky < 3; ++ky)
#pragma unroll
                for (int kx = 0; kx < 3; ++kx)
                    acc += w[(ic * 3 + ky) * 3 + kx] * patch[ic][ky][kx];
        out[oc] = (bf16_t)fmaxf(acc, 0.f);
    }
}

// ---------------------------------------------------------------------------
// 3) Repack conv weights (OC,IC,3,3) fp32 -> [OC][kpos][IC] bf16 so the WMMA
// A-fragment per lane is two contiguous 8-element K runs.
// ---------------------------------------------------------------------------
__global__ __launch_bounds__(256)
void pack_w_kernel(const float* __restrict__ w, bf16_t* __restrict__ wT) {
    const int i = blockIdx.x * 256 + threadIdx.x;
    if (i >= HID * HID * 9) return;
    const int oc = i / (HID * 9);
    const int rem = i % (HID * 9);
    const int kpos = rem / HID;
    const int ic = rem % HID;
    const int ky = kpos / 3, kx = kpos % 3;
    wT[((size_t)oc * 9 + kpos) * HID + ic] =
        (bf16_t)w[(((size_t)oc * HID + ic) * 3 + ky) * 3 + kx];
}

// ---------------------------------------------------------------------------
// 4) Implicit-GEMM 3x3 conv, 128 -> 128, ReLU, bf16 WMMA / fp32 accumulate.
// Block = 256 threads = 8 waves; each wave owns a 16-OC slice.
// Input tile (8x8 + halo, channel-fastest) staged in LDS via the CDNA5
// async memory->LDS path (GLOBAL_LOAD_ASYNC_TO_LDS_B128 / ASYNCcnt); halo
// chunks outside the image are zero-filled with plain ds stores.
// kpos is a runtime loop + ic0 unrolled x2 to stay within 256 VGPRs
// (round-0 full unroll caused scratch spills).
// ---------------------------------------------------------------------------
__global__ __launch_bounds__(256)
void conv3x3_wmma_kernel(const bf16_t* __restrict__ in, const bf16_t* __restrict__ wT,
                         const float* __restrict__ bias, bf16_t* __restrict__ out) {
    __shared__ __align__(32) bf16_t tile[10 * 10 * HID];   // 25.6 KB
    const int b = blockIdx.z;
    const int gy0 = blockIdx.y * 8;
    const int gx0 = blockIdx.x * 8;
    const int tid = threadIdx.x;

    v8bf zero8;
#pragma unroll
    for (int k = 0; k < 8; ++k) zero8[k] = (__bf16)0.0f;

    // cooperative halo load: async global->LDS for in-range 16B chunks,
    // ds zero-store for out-of-range halo chunks.
    for (int t = tid; t < 100 * 16; t += 256) {
        const int pix = t >> 4;
        const int cv = (t & 15) * 8;
        const int py = pix / 10, px = pix % 10;
        const int y = gy0 + py - 1, x = gx0 + px - 1;
        if (((unsigned)y < GRES) && ((unsigned)x < GRES)) {
            // LDS byte offset within the workgroup allocation (HW adds LDS_BASE)
            const unsigned ldsb = (unsigned)((pix * HID + cv) * sizeof(bf16_t));
            const unsigned long long ga = (unsigned long long)(uintptr_t)
                &in[(((size_t)b * GRES + y) * GRES + x) * HID + cv];
            asm volatile("global_load_async_to_lds_b128 %0, %1, off"
                         :: "v"(ldsb), "v"(ga) : "memory");
        } else {
            *(v8bf*)&tile[pix * HID + cv] = zero8;
        }
    }
    asm volatile("s_wait_asynccnt 0" ::: "memory");
    __syncthreads();

    const int lane = tid & 31;
    const int wv   = tid >> 5;
    const int ocb  = wv * 16;
    const int half = lane >> 4;   // 0: K 0..15, 1: K 16..31
    const int nl   = lane & 15;
    const int aoff0 = half ? 16 : 0;
    const int aoff1 = half ? 24 : 8;

    v8f acc[4];
#pragma unroll
    for (int n = 0; n < 4; ++n)
#pragma unroll
        for (int r = 0; r < 8; ++r) acc[n][r] = 0.f;

#pragma unroll 1
    for (int kpos = 0; kpos < 9; ++kpos) {
        const int ky = kpos / 3;
        const int kx = kpos - 3 * ky;
        const size_t arow = ((size_t)(ocb + nl) * 9 + kpos) * HID;
#pragma unroll 2
        for (int ic0 = 0; ic0 < HID; ic0 += 32) {
            // A fragment: weights row oc = ocb + nl, two contiguous K runs
            ABFrag a;
            a.h[0] = *(const v8bf*)&wT[arow + ic0 + aoff0];
            a.h[1] = *(const v8bf*)&wT[arow + ic0 + aoff1];
#pragma unroll
            for (int n = 0; n < 4; ++n) {
                const int p  = n * 16 + nl;
                const int py = (p >> 3) + ky;
                const int px = (p & 7) + kx;
                const v16bf bfrag =
                    *(const v16bf*)&tile[(py * 10 + px) * HID + ic0 + half * 16];
                acc[n] = __builtin_amdgcn_wmma_f32_16x16x32_bf16(
                    false, a.v, false, bfrag, (short)0, acc[n], false, false);
            }
        }
    }

    // epilogue: bias + ReLU + bf16 store, channel-fastest layout
#pragma unroll
    for (int n = 0; n < 4; ++n) {
        const int p  = n * 16 + nl;
        const int gy = gy0 + (p >> 3);
        const int gx = gx0 + (p & 7);
        const size_t obase = (((size_t)b * GRES + gy) * GRES + gx) * HID;
#pragma unroll
        for (int r = 0; r < 8; ++r) {
            const int oc = ocb + r + half * 8;
            const float v = acc[n][r] + bias[oc];
            out[obase + oc] = (bf16_t)fmaxf(v, 0.f);
        }
    }
}

// ---------------------------------------------------------------------------
// 5) Fold the 1x1 head (wf, bf) into conv4 (w4, b4): both linear, no ReLU
// between. Fused weights wf4: [2][kpos][IC] fp32, fused bias bf4: [2].
// ---------------------------------------------------------------------------
__global__ __launch_bounds__(256)
void fuse_wf_kernel(const float* __restrict__ w4, const float* __restrict__ b4,
                    const float* __restrict__ wf, const float* __restrict__ bfb,
                    float* __restrict__ wf4, float* __restrict__ bf4) {
    const int i = blockIdx.x * 256 + threadIdx.x;
    if (i < 2 * 9 * HID) {
        const int o = i / (9 * HID);
        const int rem = i % (9 * HID);
        const int kpos = rem / HID;
        const int ic = rem % HID;
        const int ky = kpos / 3, kx = kpos % 3;
        float s = 0.f;
#pragma unroll
        for (int m = 0; m < 4; ++m)
            s += wf[o * 4 + m] * w4[(((size_t)m * HID + ic) * 3 + ky) * 3 + kx];
        wf4[i] = s;
    }
    if (i < 2) {
        float s = bfb[i];
#pragma unroll
        for (int m = 0; m < 4; ++m) s += wf[i * 4 + m] * b4[m];
        bf4[i] = s;
    }
}

// ---------------------------------------------------------------------------
// 6) Fused conv4 ∘ wf: 128 -> 2, 3x3, no activation.
// out_grid layout: [B][H][W][2] fp32.
// ---------------------------------------------------------------------------
__global__ __launch_bounds__(256)
void out_conv_kernel(const bf16_t* __restrict__ h, const float* __restrict__ wf4,
                     const float* __restrict__ bf4, float* __restrict__ og) {
    const int b = blockIdx.y;
    const int g = blockIdx.x * 256 + threadIdx.x;
    const int row = g >> 7, col = g & 127;
    float a0 = bf4[0], a1 = bf4[1];
#pragma unroll
    for (int ky = 0; ky < 3; ++ky) {
#pragma unroll
        for (int kx = 0; kx < 3; ++kx) {
            const int y = row + ky - 1, x = col + kx - 1;
            if (((unsigned)y >= GRES) || ((unsigned)x >= GRES)) continue;
            const int kpos = ky * 3 + kx;
            const bf16_t* hp = &h[(((size_t)b * GRES + y) * GRES + x) * HID];
            const float* w0 = &wf4[(size_t)(0 * 9 + kpos) * HID];
            const float* w1 = &wf4[(size_t)(1 * 9 + kpos) * HID];
            for (int ic = 0; ic < HID; ++ic) {
                const float hv = (float)hp[ic];
                a0 += w0[ic] * hv;
                a1 += w1[ic] * hv;
            }
        }
    }
    float* o = og + ((size_t)b * GRES * GRES + g) * 2;
    o[0] = a0;
    o[1] = a1;
}

// ---------------------------------------------------------------------------
// 7) Bilinear sample (align_corners, zeros padding) + softplus head.
// d_out: mu (B*NT) then sig (B*NT), fp32.
// ---------------------------------------------------------------------------
__global__ __launch_bounds__(256)
void sample_kernel(const float* __restrict__ xt, const float* __restrict__ og_all,
                   float* __restrict__ out) {
    const int b = blockIdx.y;
    const int t = blockIdx.x * 256 + threadIdx.x;
    const float x = xt[((size_t)b * NT + t) * 2 + 0];
    const float y = xt[((size_t)b * NT + t) * 2 + 1];
    const float ix = (x + 1.f) * (float)(GRES - 1) * 0.5f;
    const float iy = (y + 1.f) * (float)(GRES - 1) * 0.5f;
    const float x0 = floorf(ix), y0 = floorf(iy);
    const float x1 = x0 + 1.f,  y1 = y0 + 1.f;
    const float wx1 = ix - x0, wx0 = 1.f - wx1;
    const float wy1 = iy - y0, wy0 = 1.f - wy1;
    const float* og = og_all + (size_t)b * GRES * GRES * 2;
    float mu = 0.f, ls = 0.f;
#define GATH(XI, YI, W)                                                        \
    do {                                                                       \
        const bool valid = ((XI) >= 0.f) && ((XI) <= (float)(GRES - 1)) &&     \
                           ((YI) >= 0.f) && ((YI) <= (float)(GRES - 1));       \
        const int xcl = (int)fminf(fmaxf((XI), 0.f), (float)(GRES - 1));       \
        const int ycl = (int)fminf(fmaxf((YI), 0.f), (float)(GRES - 1));       \
        if (valid) {                                                           \
            const float* p = og + ((size_t)ycl * GRES + xcl) * 2;              \
            mu += (W) * p[0];                                                  \
            ls += (W) * p[1];                                                  \
        }                                                                      \
    } while (0)
    GATH(x0, y0, wx0 * wy0);
    GATH(x1, y0, wx1 * wy0);
    GATH(x0, y1, wx0 * wy1);
    GATH(x1, y1, wx1 * wy1);
#undef GATH
    const float sp = (ls > 20.f) ? ls : log1pf(__expf(ls));
    out[(size_t)b * NT + t] = mu;
    out[(size_t)BATCH * NT + (size_t)b * NT + t] = 0.1f + 0.9f * sp;
}

// ---------------------------------------------------------------------------
extern "C" void kernel_launch(void* const* d_in, const int* in_sizes, int n_in,
                              void* d_out, int out_size, void* d_ws, size_t ws_size,
                              hipStream_t stream) {
    (void)in_sizes; (void)n_in; (void)out_size;
    const float* xc  = (const float*)d_in[0];
    const float* yc  = (const float*)d_in[1];
    const float* xt  = (const float*)d_in[2];
    const float* sig = (const float*)d_in[3];
    const float* w1  = (const float*)d_in[4];
    const float* b1  = (const float*)d_in[5];
    const float* w2  = (const float*)d_in[6];
    const float* b2  = (const float*)d_in[7];
    const float* w3  = (const float*)d_in[8];
    const float* b3  = (const float*)d_in[9];
    const float* w4  = (const float*)d_in[10];
    const float* b4  = (const float*)d_in[11];
    const float* wf  = (const float*)d_in[12];
    const float* bfb = (const float*)d_in[13];
    float* out = (float*)d_out;

    char* ws = (char*)d_ws;
    size_t off = 0;
    auto alloc = [&](size_t bytes) -> char* {
        char* p = ws + off;
        off += (bytes + 255) & ~(size_t)255;
        return p;
    };
    const size_t actB = (size_t)BATCH * GRES * GRES * HID * sizeof(bf16_t); // 32 MB
    float*  feat = (float*) alloc((size_t)BATCH * GRES * GRES * 2 * sizeof(float));
    bf16_t* h1   = (bf16_t*)alloc(actB);
    bf16_t* h2   = (bf16_t*)alloc(actB);
    bf16_t* wT2  = (bf16_t*)alloc((size_t)HID * HID * 9 * sizeof(bf16_t));
    bf16_t* wT3  = (bf16_t*)alloc((size_t)HID * HID * 9 * sizeof(bf16_t));
    float*  wf4  = (float*) alloc((size_t)2 * 9 * HID * sizeof(float));
    float*  bf4  = (float*) alloc(256);
    float*  og   = (float*) alloc((size_t)BATCH * GRES * GRES * 2 * sizeof(float));
    bf16_t* h3   = h1;   // h1 is dead after conv2; reuse for conv3 output
    if (off > ws_size) return;   // workspace too small: no-op

    const dim3 blk(256);
    rbf_kernel  <<<dim3(GRES * GRES / 256, BATCH), blk, 0, stream>>>(xc, yc, sig, feat);
    conv1_kernel<<<dim3(GRES * GRES / 256, BATCH), blk, 0, stream>>>(feat, w1, b1, h1);
    pack_w_kernel<<<dim3((HID * HID * 9 + 255) / 256), blk, 0, stream>>>(w2, wT2);
    pack_w_kernel<<<dim3((HID * HID * 9 + 255) / 256), blk, 0, stream>>>(w3, wT3);
    conv3x3_wmma_kernel<<<dim3(GRES / 8, GRES / 8, BATCH), blk, 0, stream>>>(h1, wT2, b2, h2);
    conv3x3_wmma_kernel<<<dim3(GRES / 8, GRES / 8, BATCH), blk, 0, stream>>>(h2, wT3, b3, h3);
    fuse_wf_kernel<<<dim3((2 * 9 * HID + 255) / 256), blk, 0, stream>>>(w4, b4, wf, bfb, wf4, bf4);
    out_conv_kernel<<<dim3(GRES * GRES / 256, BATCH), blk, 0, stream>>>(h3, wf4, bf4, og);
    sample_kernel<<<dim3(NT / 256, BATCH), blk, 0, stream>>>(xt, og, out);
}
